// LSTMClassifier_82197084111358
// MI455X (gfx1250) — matrix operand order repaired and checked by reading, not compiled, not run on hardware
//
#include <hip/hip_runtime.h>
#include <hip/hip_bf16.h>

// ---------------------------------------------------------------------------
// LSTMClassifier for MI455X (gfx1250): 36 independent 2-layer LSTMs (H=256)
// over T=128 steps, batch 32, followed by a small MLP head.
//
//  * 232 GFLOP recurrence; weights (108MB f32 -> 54MB f16) stay L2-resident:
//    packed once into WMMA-B-fragment order, streamed from L2 every step.
//  * v_wmma_f32_16x16x32_f16 (f16 in, f32 accumulate; h in (-1,1), safe).
//  * One workgroup per stream (36 WGs, 8 waves), persistent over 128 steps;
//    h1/h2 in LDS (f16, padded stride 264), c1/c2 in fragment-layout VGPRs.
//  * Gates processed sequentially (g~, i, f, o) with algebraic folding so a
//    wave never holds more than 4 accumulator tiles -> no spills.
//  * NOTE: k-loop kept as a uniform 8-iteration loop with partial unroll 2.
//    (Peeling kt=0 for an inline-zero C operand made LLVM fully unroll the
//    gate and spill B fragments to scratch -- reverted.)
// ---------------------------------------------------------------------------

typedef __attribute__((ext_vector_type(16))) _Float16 v16h;
typedef __attribute__((ext_vector_type(8)))  _Float16 v8h;
typedef __attribute__((ext_vector_type(8)))  float    v8f;

#define NS   36          // streams
#define HH   256         // hidden
#define GG   1024        // 4*H gate width
#define BB   32          // batch
#define TT   128         // timesteps
#define HS   264         // padded LDS row stride (f16 elems)
#define KF   9472        // (NS+1)*HH feature width for the head
#define KT1  296         // KF/32 k-tiles for head GEMM
#define NOUT 30

// ---- math helpers ---------------------------------------------------------
static __device__ __forceinline__ float sigm(float x) {
  return 1.0f / (1.0f + __expf(-x));
}
static __device__ __forceinline__ float tanh_(float x) {
  return 1.0f - 2.0f / (__expf(2.0f * x) + 1.0f);
}

// A-fragment (16x32 f16, ISA 7.12.2): lane L holds row M=L%16; f16 pairs at
// K = 8*(L/16)+{0..7} (elems 0..7) and K = 16+8*(L/16)+{0..7} (elems 8..15).
static __device__ __forceinline__ v16h make_a(const _Float16* p) {
  v8h lo = *(const v8h*)(p);
  v8h hi = *(const v8h*)(p + 16);
  return __builtin_shufflevector(lo, hi, 0, 1, 2, 3, 4, 5, 6, 7,
                                 8, 9, 10, 11, 12, 13, 14, 15);
}

static __device__ __forceinline__ v8f wmma_f16(v16h a, v16h b, v8f c) {
  return __builtin_amdgcn_wmma_f32_16x16x32_f16(false, a, false, b,
                                                (short)0, c, false, false);
}

// ---------------------------------------------------------------------------
// Pack kernels: rewrite weights as f16 B-fragments.
// Block (gt,kt) = 16 gate-cols x 32 k: 1024 B; lane L owns bytes [32L,32L+32).
// Element e of lane L: col = gt*16 + (L&15), k = kt*32 + 8*(L>>4) + (e&7) + 16*(e>>3).
// ---------------------------------------------------------------------------
__global__ void pack_lstm_weights(const float* __restrict__ Whh1,
                                  const float* __restrict__ Wih2,
                                  const float* __restrict__ Whh2,
                                  _Float16* __restrict__ pack) {
  long idx = (long)blockIdx.x * blockDim.x + threadIdx.x;  // < 28,311,552
  int e = (int)(idx & 15);
  int L = (int)((idx >> 4) & 31);
  long blk = idx >> 9;                 // ((n*3+m)*64 + gt)*8 + kt
  int kt = (int)(blk & 7);
  long r = blk >> 3;
  int gt = (int)(r & 63); r >>= 6;
  int m = (int)(r % 3);
  int n = (int)(r / 3);
  int g = gt * 16 + (L & 15);
  int k = kt * 32 + ((L >> 4) << 3) + (e & 7) + ((e >> 3) << 4);
  const float* W = (m == 0) ? Whh1 : (m == 1) ? Wih2 : Whh2;
  float v = W[((long)n * GG + g) * HH + k];   // W[n][g][h=k]
  pack[blk * 512 + L * 16 + e] = (_Float16)v;
}

__global__ void pack_wp1_kernel(const float* __restrict__ Wp1,
                                _Float16* __restrict__ pack) {
  int idx = blockIdx.x * blockDim.x + threadIdx.x;  // < 2,424,832
  int e = idx & 15;
  int L = (idx >> 4) & 31;
  int blk = idx >> 9;                 // gt*KT1 + kt
  int kt = blk % KT1;
  int gt = blk / KT1;
  int col = gt * 16 + (L & 15);
  int k = kt * 32 + ((L >> 4) << 3) + (e & 7) + ((e >> 3) << 4);
  pack[(long)blk * 512 + L * 16 + e] = (_Float16)Wp1[(long)k * HH + col];
}

__global__ void fuse_bias(const float* __restrict__ bih1, const float* __restrict__ bhh1,
                          const float* __restrict__ bih2, const float* __restrict__ bhh2,
                          float* __restrict__ bg1, float* __restrict__ bg2) {
  int i = blockIdx.x * blockDim.x + threadIdx.x;
  if (i < NS * GG) {
    bg1[i] = bih1[i] + bhh1[i];
    bg2[i] = bih2[i] + bhh2[i];
  }
}

__global__ void spec_dense(const float* __restrict__ spec, const float* __restrict__ Wspec,
                           const float* __restrict__ bspec, _Float16* __restrict__ hidden) {
  int i = blockIdx.x * blockDim.x + threadIdx.x;  // < 8192
  int b = i >> 8, hcol = i & 255;
  float s = bspec[hcol];
#pragma unroll
  for (int k = 0; k < 6; ++k) s += spec[b * 6 + k] * Wspec[k * HH + hcol];
  hidden[(long)b * KF + NS * HH + hcol] = (_Float16)s;
}

// ---------------------------------------------------------------------------
// One gate k-sweep: acc[ct][bt] += A(src) x B(pm, gate-tile gt0+ct).
// K=256 -> 8 k-tiles; partial unroll keeps ~8 loads in flight (no spills).
// ---------------------------------------------------------------------------
static __device__ __forceinline__ void
gate_mm(const _Float16* __restrict__ src, const _Float16* __restrict__ pm,
        int gt0, int lane, v8f acc[2][2]) {
  const int half = lane >> 4, lc = lane & 15;
#pragma unroll 2
  for (int kt = 0; kt < 8; ++kt) {
    const int kb = kt * 32 + 8 * half;
    v16h a0 = make_a(&src[lc * HS + kb]);
    v16h a1 = make_a(&src[(16 + lc) * HS + kb]);
#pragma unroll
    for (int ct = 0; ct < 2; ++ct) {
      const _Float16* bp = pm + ((long)(gt0 + ct) * 8 + kt) * 512 + (long)lane * 16;
      if (ct == 0 && kt < 7) __builtin_prefetch(bp + 512, 0, 1);  // next k-tile
      v16h bf = *(const v16h*)bp;
      acc[ct][0] = wmma_f16(a0, bf, acc[ct][0]);
      acc[ct][1] = wmma_f16(a1, bf, acc[ct][1]);
    }
  }
}

static __device__ __forceinline__ void zero4(v8f a[2][2]) {
  const v8f z = {};
  a[0][0] = z; a[0][1] = z; a[1][0] = z; a[1][1] = z;
}

// ---------------------------------------------------------------------------
// Persistent LSTM kernel: one block per stream, 256 threads (8 waves).
// Wave w owns h-columns [w*32, w*32+32) of both layers.
// ---------------------------------------------------------------------------
__global__ __launch_bounds__(256, 1)
void lstm_persistent(const float* __restrict__ x, const float* __restrict__ Wih1,
                     const _Float16* __restrict__ pack, const float* __restrict__ bg1,
                     const float* __restrict__ bg2, _Float16* __restrict__ hidden) {
  __shared__ __align__(16) _Float16 h1a[BB * HS];
  __shared__ __align__(16) _Float16 h1b[BB * HS];
  __shared__ __align__(16) _Float16 h2s[BB * HS];
  __shared__ float xbuf[BB];

  const int n = blockIdx.x;
  const int tid = threadIdx.x;
  const int lane = tid & 31;
  const int w = tid >> 5;
  const int half = lane >> 4;
  const int lc = lane & 15;
  const int col0 = w * 32 + lc;
  const int gtw = w * 2;               // wave's first gate-coltile offset

  for (int i = tid; i < BB * HS; i += 256) {
    h1a[i] = (_Float16)0.0f;
    h2s[i] = (_Float16)0.0f;
  }

  // Per-lane column constants (col depends only on lane%16).
  float bgl1[4][2], bgl2[4][2], w1l[4][2];
#pragma unroll
  for (int g = 0; g < 4; ++g) {
#pragma unroll
    for (int ct = 0; ct < 2; ++ct) {
      int gc = g * HH + col0 + ct * 16;
      bgl1[g][ct] = bg1[n * GG + gc];
      bgl2[g][ct] = bg2[n * GG + gc];
      w1l[g][ct] = Wih1[n * GG + gc];
    }
  }

  v8f c1r[2][2] = {}, c2r[2][2] = {};
  const _Float16* packN = pack + (long)n * 3 * 64 * 8 * 512;
  const _Float16* pm0 = packN;                       // Whh1
  const _Float16* pm1 = packN + (long)64 * 8 * 512;  // Wih2
  const _Float16* pm2 = pm1 + (long)64 * 8 * 512;    // Whh2

  for (int t = 0; t < TT; ++t) {
    _Float16* cur = (t & 1) ? h1b : h1a;
    _Float16* nxt = (t & 1) ? h1a : h1b;
    if (tid < BB) xbuf[tid] = x[((long)tid * TT + t) * NS + n];
    __syncthreads();  // xbuf visible; prior-step h2 writes visible

    v8f acc[2][2], tmp[2][2], hv[2][2];

    // ======== layer 1: gates sequenced g~, i, f, o (reads cur) ============
    // g~ : tmp = tanh(g~)
    zero4(acc);
    gate_mm(cur, pm0, 2 * 16 + gtw, lane, acc);
#pragma unroll
    for (int ct = 0; ct < 2; ++ct)
#pragma unroll
      for (int bt = 0; bt < 2; ++bt)
#pragma unroll
        for (int e = 0; e < 8; ++e) {
          float xv = xbuf[bt * 16 + e + 8 * half];
          tmp[ct][bt][e] = tanh_(acc[ct][bt][e] + bgl1[2][ct] + xv * w1l[2][ct]);
        }
    // i : tmp = sigm(i) * tanh(g~)
    zero4(acc);
    gate_mm(cur, pm0, 0 * 16 + gtw, lane, acc);
#pragma unroll
    for (int ct = 0; ct < 2; ++ct)
#pragma unroll
      for (int bt = 0; bt < 2; ++bt)
#pragma unroll
        for (int e = 0; e < 8; ++e) {
          float xv = xbuf[bt * 16 + e + 8 * half];
          tmp[ct][bt][e] *= sigm(acc[ct][bt][e] + bgl1[0][ct] + xv * w1l[0][ct]);
        }
    // f : c1 = sigm(f)*c1 + tmp
    zero4(acc);
    gate_mm(cur, pm0, 1 * 16 + gtw, lane, acc);
#pragma unroll
    for (int ct = 0; ct < 2; ++ct)
#pragma unroll
      for (int bt = 0; bt < 2; ++bt)
#pragma unroll
        for (int e = 0; e < 8; ++e) {
          float xv = xbuf[bt * 16 + e + 8 * half];
          c1r[ct][bt][e] = sigm(acc[ct][bt][e] + bgl1[1][ct] + xv * w1l[1][ct]) *
                               c1r[ct][bt][e] + tmp[ct][bt][e];
        }
    // o : hv = sigm(o)*tanh(c1)
    zero4(acc);
    gate_mm(cur, pm0, 3 * 16 + gtw, lane, acc);
#pragma unroll
    for (int ct = 0; ct < 2; ++ct)
#pragma unroll
      for (int bt = 0; bt < 2; ++bt)
#pragma unroll
        for (int e = 0; e < 8; ++e) {
          float xv = xbuf[bt * 16 + e + 8 * half];
          hv[ct][bt][e] = sigm(acc[ct][bt][e] + bgl1[3][ct] + xv * w1l[3][ct]) *
                          tanh_(c1r[ct][bt][e]);
        }
    __syncthreads();  // all reads of cur finished
#pragma unroll
    for (int ct = 0; ct < 2; ++ct)
#pragma unroll
      for (int bt = 0; bt < 2; ++bt)
#pragma unroll
        for (int e = 0; e < 8; ++e)
          nxt[(bt * 16 + e + 8 * half) * HS + col0 + ct * 16] =
              (_Float16)hv[ct][bt][e];
    __syncthreads();  // h1_new visible

    // ======== layer 2: gates from h1_new @ Wih2 + h2_old @ Whh2 ===========
    // g~
    zero4(acc);
    gate_mm(nxt, pm1, 2 * 16 + gtw, lane, acc);
    gate_mm(h2s, pm2, 2 * 16 + gtw, lane, acc);
#pragma unroll
    for (int ct = 0; ct < 2; ++ct)
#pragma unroll
      for (int bt = 0; bt < 2; ++bt)
#pragma unroll
        for (int e = 0; e < 8; ++e)
          tmp[ct][bt][e] = tanh_(acc[ct][bt][e] + bgl2[2][ct]);
    // i
    zero4(acc);
    gate_mm(nxt, pm1, 0 * 16 + gtw, lane, acc);
    gate_mm(h2s, pm2, 0 * 16 + gtw, lane, acc);
#pragma unroll
    for (int ct = 0; ct < 2; ++ct)
#pragma unroll
      for (int bt = 0; bt < 2; ++bt)
#pragma unroll
        for (int e = 0; e < 8; ++e)
          tmp[ct][bt][e] *= sigm(acc[ct][bt][e] + bgl2[0][ct]);
    // f
    zero4(acc);
    gate_mm(nxt, pm1, 1 * 16 + gtw, lane, acc);
    gate_mm(h2s, pm2, 1 * 16 + gtw, lane, acc);
#pragma unroll
    for (int ct = 0; ct < 2; ++ct)
#pragma unroll
      for (int bt = 0; bt < 2; ++bt)
#pragma unroll
        for (int e = 0; e < 8; ++e)
          c2r[ct][bt][e] = sigm(acc[ct][bt][e] + bgl2[1][ct]) * c2r[ct][bt][e] +
                           tmp[ct][bt][e];
    // o
    zero4(acc);
    gate_mm(nxt, pm1, 3 * 16 + gtw, lane, acc);
    gate_mm(h2s, pm2, 3 * 16 + gtw, lane, acc);
#pragma unroll
    for (int ct = 0; ct < 2; ++ct)
#pragma unroll
      for (int bt = 0; bt < 2; ++bt)
#pragma unroll
        for (int e = 0; e < 8; ++e)
          hv[ct][bt][e] = sigm(acc[ct][bt][e] + bgl2[3][ct]) *
                          tanh_(c2r[ct][bt][e]);
    __syncthreads();  // all reads of h2_old finished
#pragma unroll
    for (int ct = 0; ct < 2; ++ct)
#pragma unroll
      for (int bt = 0; bt < 2; ++bt)
#pragma unroll
        for (int e = 0; e < 8; ++e)
          h2s[(bt * 16 + e + 8 * half) * HS + col0 + ct * 16] =
              (_Float16)hv[ct][bt][e];
  }

  // -------- export final h2 into the concatenated feature buffer ----------
  __syncthreads();
  for (int i = tid; i < BB * HH; i += 256) {
    int b = i >> 8, col = i & 255;
    hidden[(long)b * KF + n * HH + col] = h2s[b * HS + col];
  }
}

// ---------------------------------------------------------------------------
// Head: t1 = relu(feat[32,9472] @ Wp1 + bp1); out = t1 @ Wp2 + bp2. 1 WG.
// ---------------------------------------------------------------------------
__global__ __launch_bounds__(256, 1)
void mlp_head(const _Float16* __restrict__ hidden, const _Float16* __restrict__ packW,
              const float* __restrict__ bp1, const float* __restrict__ Wp2,
              const float* __restrict__ bp2, float* __restrict__ out) {
  __shared__ __align__(16) float t1[BB * HS];
  const int tid = threadIdx.x;
  const int lane = tid & 31;
  const int w = tid >> 5;
  const int half = lane >> 4, lc = lane & 15;

  v8f acc[2][2] = {};
#pragma unroll 2
  for (int kt = 0; kt < KT1; ++kt) {
    const int kb = kt * 32 + 8 * half;
    v16h a0 = make_a(&hidden[(long)lc * KF + kb]);
    v16h a1 = make_a(&hidden[(long)(16 + lc) * KF + kb]);
#pragma unroll
    for (int ct = 0; ct < 2; ++ct) {
      const int gt = w * 2 + ct;
      v16h bf = *(const v16h*)(packW + ((long)gt * KT1 + kt) * 512 + (long)lane * 16);
      acc[ct][0] = wmma_f16(a0, bf, acc[ct][0]);
      acc[ct][1] = wmma_f16(a1, bf, acc[ct][1]);
    }
  }
#pragma unroll
  for (int ct = 0; ct < 2; ++ct) {
    float bias = bp1[w * 32 + ct * 16 + lc];
#pragma unroll
    for (int bt = 0; bt < 2; ++bt) {
#pragma unroll
      for (int e = 0; e < 8; ++e) {
        int b = bt * 16 + e + 8 * half;
        float v = acc[ct][bt][e] + bias;
        t1[b * HS + w * 32 + ct * 16 + lc] = fmaxf(v, 0.0f);
      }
    }
  }
  __syncthreads();
  for (int oi = tid; oi < BB * NOUT; oi += 256) {
    int b = oi / NOUT, oc = oi % NOUT;
    float s = bp2[oc];
    for (int k = 0; k < HH; ++k) s += t1[b * HS + k] * Wp2[k * NOUT + oc];
    out[oi] = s;
  }
}

// ---------------------------------------------------------------------------
extern "C" void kernel_launch(void* const* d_in, const int* in_sizes, int n_in,
                              void* d_out, int out_size, void* d_ws, size_t ws_size,
                              hipStream_t stream) {
  (void)in_sizes; (void)n_in; (void)out_size; (void)ws_size;
  const float* x     = (const float*)d_in[0];
  const float* spec  = (const float*)d_in[1];
  const float* Wih1  = (const float*)d_in[2];
  const float* Whh1  = (const float*)d_in[3];
  const float* bih1  = (const float*)d_in[4];
  const float* bhh1  = (const float*)d_in[5];
  const float* Wih2  = (const float*)d_in[6];
  const float* Whh2  = (const float*)d_in[7];
  const float* bih2  = (const float*)d_in[8];
  const float* bhh2  = (const float*)d_in[9];
  const float* Wspec = (const float*)d_in[10];
  const float* bspec = (const float*)d_in[11];
  const float* Wp1   = (const float*)d_in[12];
  const float* bp1   = (const float*)d_in[13];
  const float* Wp2   = (const float*)d_in[14];
  const float* bp2   = (const float*)d_in[15];
  float* out = (float*)d_out;

  // workspace layout (bytes), all blocks 32B-aligned
  char* ws = (char*)d_ws;
  _Float16* packL  = (_Float16*)(ws);                // 36*3*64*8*1024 = 56,623,104
  _Float16* packP  = (_Float16*)(ws + 56623104);     // 16*296*1024   =  4,849,664
  float*    bg1    = (float*)(ws + 61472768);        // 147,456
  float*    bg2    = (float*)(ws + 61620224);        // 147,456
  _Float16* hidden = (_Float16*)(ws + 61767680);     // 606,208  (total ~59.5 MB)

  pack_lstm_weights<<<110592, 256, 0, stream>>>(Whh1, Wih2, Whh2, packL);
  pack_wp1_kernel<<<9472, 256, 0, stream>>>(Wp1, packP);
  fuse_bias<<<144, 256, 0, stream>>>(bih1, bhh1, bih2, bhh2, bg1, bg2);
  spec_dense<<<32, 256, 0, stream>>>(spec, Wspec, bspec, hidden);
  lstm_persistent<<<NS, 256, 0, stream>>>(x, Wih1, packL, bg1, bg2, hidden);
  mlp_head<<<1, 256, 0, stream>>>(hidden, packP, bp1, Wp2, bp2, out);
}
